// DecoderLayer_38817914421606
// MI455X (gfx1250) — matrix operand order repaired and checked by reading, not compiled, not run on hardware
//
#include <hip/hip_runtime.h>
#include <cstdint>
#include <cstddef>

typedef __bf16 bf16;
typedef __attribute__((ext_vector_type(16))) __bf16 v16bf;
typedef __attribute__((ext_vector_type(8)))  __bf16 v8bf;
typedef __attribute__((ext_vector_type(8)))  float  v8f;
typedef __attribute__((ext_vector_type(4)))  unsigned int v4u;
typedef __attribute__((ext_vector_type(8)))  int v8i;
typedef __attribute__((ext_vector_type(4)))  int v4i;

#define D_MODEL  2048
#define N_HEADS  16
#define HEAD_DIM 128
#define D_FF     8192
#define BATCH    2
#define SEQ      2048
#define ROWS     (BATCH * SEQ)   // 4096
#define LN_EPS   1e-5f

// ---------------------------------------------------------------- helpers

__device__ inline v8f zero8() {
  v8f z;
#pragma unroll
  for (int i = 0; i < 8; ++i) z[i] = 0.f;
  return z;
}

// Build a 16-element bf16 fragment from two 16-byte chunks (ds_load_b128 pair).
__device__ inline v16bf ldfrag(const bf16* p0, const bf16* p1) {
  union { v16bf v; v8bf h[2]; } u;
  u.h[0] = *(const v8bf*)p0;
  u.h[1] = *(const v8bf*)p1;
  return u.v;
}

__device__ inline v8f wmma_bf16(v16bf a, v16bf b, v8f c) {
  // D = A(16x32 bf16) * B(32x16 bf16) + C(16x16 f32)
  return __builtin_amdgcn_wmma_f32_16x16x32_bf16(
      /*neg_a=*/false, a, /*neg_b=*/false, b,
      /*c_mod=*/(short)0, c, /*reuse_a=*/false, /*reuse_b=*/false);
}

// ------------------------------------------------ Tensor Data Mover (TDM)
// DMA a [tile_rows x 32] bf16 tile (row stride = stride_elems in global,
// 40 elements in LDS via pad fields) straight into LDS. Issued by one wave;
// completion tracked with TENSORcnt.
__device__ inline void tdm_load_tile_bf16(
    unsigned int lds_off, const bf16* gaddr, int tensor_d0, long stride_elems) {
  const unsigned long long ga = (unsigned long long)(uintptr_t)gaddr;
  v4u g0;
  g0[0] = 1u;                                    // count=1 (valid), user mode
  g0[1] = lds_off;                               // lds_addr (bytes)
  g0[2] = (unsigned int)(ga & 0xffffffffu);      // global_addr[31:0]
  g0[3] = (unsigned int)((ga >> 32) & 0x01ffffffu) | (2u << 30);  // [56:32]+type=2

  const unsigned long long t0 = (unsigned int)tensor_d0;   // tensor_dim0 = K
  const unsigned long long t1 = 1u << 20;                  // tensor_dim1 (ample)
  const unsigned long long s0 = (unsigned long long)stride_elems;  // dim0 stride
  v8i g1;
  g1[0] = (int)((1u << 16)      // data_size = 1 -> 2 bytes
              | (1u << 20)      // pad_enable
              | (3u << 22)      // pad_interval: 16 DWORDs (64B) between pads
              | (3u << 25));    // pad_amount: 4 DWORDs (16B) -> 80B LDS rows
  g1[1] = (int)((t0 & 0xffffu) << 16);                       // tensor_dim0[15:0]
  g1[2] = (int)(((t0 >> 16) & 0xffffu) | ((t1 & 0xffffu) << 16));
  g1[3] = (int)(((t1 >> 16) & 0xffffu) | (32u << 16));       // tile_dim0 = 32
  g1[4] = (int)(128u);                                       // tile_dim1=128, tile_dim2=0
  g1[5] = (int)(unsigned int)(s0 & 0xffffffffu);             // dim0_stride[31:0]
  g1[6] = (int)(unsigned int)((s0 >> 32) & 0xffffu);         // dim0_stride[47:32]
  g1[7] = 0;                                                 // dim1_stride = 0

  v4i z4 = {0, 0, 0, 0};
#if __clang_major__ >= 23
  v8i z8 = {0, 0, 0, 0, 0, 0, 0, 0};
  __builtin_amdgcn_tensor_load_to_lds(g0, g1, z4, z4, z8, 0);
#else
  __builtin_amdgcn_tensor_load_to_lds(g0, g1, z4, z4, 0);
#endif
}

// ---------------------------------------------- weight convert + transpose
// W [K][N] fp32 -> Wt [N][K] bf16, LDS-tiled so both sides are coalesced.
// grid: (N/32, K/32), block 256 (32 x 8).
__global__ __launch_bounds__(256) void transpose_cvt_bf16(
    const float* __restrict__ W, bf16* __restrict__ Wt, int K, int N) {
  __shared__ float tile[32][33];
  const int tx = threadIdx.x & 31;
  const int ty = threadIdx.x >> 5;           // 0..7
  const int bn = blockIdx.x * 32;
  const int bk = blockIdx.y * 32;
#pragma unroll
  for (int r = 0; r < 32; r += 8)            // coalesced read along N
    tile[ty + r][tx] = W[(size_t)(bk + ty + r) * N + bn + tx];
  __syncthreads();
#pragma unroll
  for (int r = 0; r < 32; r += 8)            // coalesced write along K
    Wt[(size_t)(bn + ty + r) * K + bk + tx] = (bf16)tile[tx][ty + r];
}

// --------------------------------------------------------- LayerNorm->bf16
__global__ __launch_bounds__(256) void layernorm_bf16(
    const float* __restrict__ x, const float* __restrict__ g,
    const float* __restrict__ bvec, bf16* __restrict__ out) {
  const int row = blockIdx.x;
  const int tid = threadIdx.x;
  const float* xr = x + (size_t)row * D_MODEL;
  float vreg[8];
  float s = 0.f, s2 = 0.f;
#pragma unroll
  for (int j = 0; j < 8; ++j) {              // single global pass, keep in regs
    float v = xr[tid + j * 256];
    vreg[j] = v; s += v; s2 += v * v;
  }
#pragma unroll
  for (int m = 16; m >= 1; m >>= 1) {
    s  += __shfl_xor(s,  m, 32);
    s2 += __shfl_xor(s2, m, 32);
  }
  __shared__ float sa[8], sb[8];
  const int wave = tid >> 5, lane = tid & 31;
  if (lane == 0) { sa[wave] = s; sb[wave] = s2; }
  __syncthreads();
  float ts = 0.f, ts2 = 0.f;
#pragma unroll
  for (int w = 0; w < 8; ++w) { ts += sa[w]; ts2 += sb[w]; }
  const float mean = ts * (1.f / D_MODEL);
  const float var  = ts2 * (1.f / D_MODEL) - mean * mean;
  const float inv  = rsqrtf(var + LN_EPS);
  bf16* orow = out + (size_t)row * D_MODEL;
#pragma unroll
  for (int j = 0; j < 8; ++j) {
    const int i = tid + j * 256;
    orow[i] = (bf16)((vreg[j] - mean) * inv * g[i] + bvec[i]);
  }
}

// ----------------------------------------------------------- WMMA GEMM
// C[M,N] = act(scale * A[M,K] @ Bt[N,K]^T + bias) + residual
// Block tile 128x128, BK=32, 8 waves (2x4), wave tile 64x32 (4x2 WMMA tiles).
// TDM double buffering: wave 0 DMAs tile t+1 into the other LDS buffer
// (TENSOR_LOAD_TO_LDS with pad -> 40-element rows) while all waves run WMMAs
// on tile t; s_wait_tensorcnt + barrier closes each step.
__global__ __launch_bounds__(256) void gemm_bf16_wmma(
    const bf16* __restrict__ A, const bf16* __restrict__ Bt,
    int M, int N, int K, float scale,
    const float* __restrict__ bias, const float* __restrict__ residual,
    int act, bf16* __restrict__ out_bf, float* __restrict__ out_f) {
  __shared__ __align__(16) bf16 As[2][128][40];   // +8 pad, 16B-aligned rows
  __shared__ __align__(16) bf16 Bs[2][128][40];

  const int tid  = threadIdx.x;
  const int wave = tid >> 5, lane = tid & 31;
  const int wm = wave >> 2, wn = wave & 3;
  const int lm = lane & 15, hi = lane >> 4;
  const int bm = blockIdx.y * 128, bn = blockIdx.x * 128;

  const bf16* atile = A  + (size_t)bm * K;   // 128 rows, stride K
  const bf16* btile = Bt + (size_t)bn * K;

  const unsigned int ldsA0 = (unsigned int)(uintptr_t)&As[0][0][0];
  const unsigned int ldsA1 = (unsigned int)(uintptr_t)&As[1][0][0];
  const unsigned int ldsB0 = (unsigned int)(uintptr_t)&Bs[0][0][0];
  const unsigned int ldsB1 = (unsigned int)(uintptr_t)&Bs[1][0][0];

  v8f acc[4][2];
#pragma unroll
  for (int mt = 0; mt < 4; ++mt)
#pragma unroll
    for (int nt = 0; nt < 2; ++nt) acc[mt][nt] = zero8();

  const int nk = K >> 5;
  if (wave == 0) {  // prologue: DMA tile 0 into buffer 0
    tdm_load_tile_bf16(ldsA0, atile, K, K);
    tdm_load_tile_bf16(ldsB0, btile, K, K);
    __builtin_amdgcn_s_wait_tensorcnt(0);
  }
  __syncthreads();

  for (int t = 0; t < nk; ++t) {
    const int cur = t & 1;
    const bool nxt = (t + 1) < nk;
    if (wave == 0 && nxt) {       // DMA tile t+1 into the other buffer
      const int k1 = (t + 1) << 5;
      tdm_load_tile_bf16(cur ? ldsA0 : ldsA1, atile + k1, K, K);
      tdm_load_tile_bf16(cur ? ldsB0 : ldsB1, btile + k1, K, K);
    }

    v16bf af[4], bfr[2];
#pragma unroll
    for (int mt = 0; mt < 4; ++mt) {
      const bf16* r = &As[cur][wm * 64 + mt * 16 + lm][0];
      af[mt] = ldfrag(r + hi * 8, r + 16 + hi * 8);      // A 16x32 layout
    }
#pragma unroll
    for (int nt = 0; nt < 2; ++nt) {
      const bf16* r = &Bs[cur][wn * 32 + nt * 16 + lm][0];
      bfr[nt] = ldfrag(r + hi * 8, r + 16 + hi * 8);     // B 32x16 layout
    }
#pragma unroll
    for (int mt = 0; mt < 4; ++mt)
#pragma unroll
      for (int nt = 0; nt < 2; ++nt)
        acc[mt][nt] = wmma_bf16(af[mt], bfr[nt], acc[mt][nt]);

    if (wave == 0 && nxt) __builtin_amdgcn_s_wait_tensorcnt(0);
    __syncthreads();
  }

  // Epilogue: C layout — VGPR i -> row i+8*hi, lane -> col lm
#pragma unroll
  for (int nt = 0; nt < 2; ++nt) {
    const int col = bn + wn * 32 + nt * 16 + lm;
    const float bi = bias ? bias[col] : 0.f;
#pragma unroll
    for (int mt = 0; mt < 4; ++mt) {
#pragma unroll
      for (int i = 0; i < 8; ++i) {
        const int row = bm + wm * 64 + mt * 16 + i + 8 * hi;
        float vv = acc[mt][nt][i] * scale + bi;
        if (act == 1)  // exact GELU
          vv = 0.5f * vv * (1.f + erff(vv * 0.70710678118654752f));
        const size_t o = (size_t)row * N + col;
        if (residual) vv += residual[o];
        if (out_f) out_f[o] = vv;
        else       out_bf[o] = (bf16)vv;
      }
    }
  }
}

// ---------------------------------------------------- flash attention
// grid (SEQ/128, N_HEADS, BATCH), 256 threads. Wave w owns Q rows
// [q0 + 16w, q0 + 16w + 16). Online softmax over 64-key tiles.
__global__ __launch_bounds__(256) void flash_attn_wmma(
    const bf16* __restrict__ q, const bf16* __restrict__ k,
    const bf16* __restrict__ v, const float* __restrict__ alibi,
    bf16* __restrict__ out) {
  __shared__ __align__(16) bf16 Kl[64][136];    // K tile, row-major [key][d]
  __shared__ __align__(16) bf16 Vt[128][72];    // V tile transposed [d][key]
  __shared__ __align__(16) bf16 Ps[8][16][72];  // per-wave P staging

  const int q0 = blockIdx.x * 128;
  const int h  = blockIdx.y;
  const int b  = blockIdx.z;
  const int tid = threadIdx.x, wave = tid >> 5, lane = tid & 31;
  const int lm = lane & 15, hi = lane >> 4;
  const float NEG = -3.40282347e38f;

  // Q fragments held in registers (already scaled by 1/sqrt(d) in Q GEMM)
  v16bf qf[4];
  {
    const int qrow = q0 + wave * 16 + lm;
    const bf16* qb = q + ((size_t)b * SEQ + qrow) * D_MODEL + h * HEAD_DIM;
#pragma unroll
    for (int c = 0; c < 4; ++c)
      qf[c] = ldfrag(qb + c * 32 + hi * 8, qb + c * 32 + 16 + hi * 8);
  }

  v8f o[8];
#pragma unroll
  for (int ot = 0; ot < 8; ++ot) o[ot] = zero8();
  float mrow[8], lsum[8];
#pragma unroll
  for (int i = 0; i < 8; ++i) { mrow[i] = NEG; lsum[i] = 0.f; }

  const int kr  = tid >> 2;                    // 0..63 (key row)
  const int kc  = (tid & 3) * 32;              // d chunk
  const size_t kvbase = (size_t)b * SEQ * D_MODEL + (size_t)h * HEAD_DIM;

  const int nkt = (q0 + 128) >> 6;  // causal: key tiles fully/partially visible
  for (int t = 0; t < nkt; ++t) {
    const int kb = t * 64;
    __syncthreads();
    if (t + 1 < nkt) {                         // prefetch next K/V tile
      __builtin_prefetch(k + kvbase + (size_t)(kb + 64 + kr) * D_MODEL + kc, 0, 1);
      __builtin_prefetch(v + kvbase + (size_t)(kb + 64 + kr) * D_MODEL + kc, 0, 1);
    }
    {  // K tile -> LDS (row major), 32 elems / thread
      const uint4* g = (const uint4*)(k + kvbase + (size_t)(kb + kr) * D_MODEL + kc);
      uint4 g0 = g[0], g1 = g[1], g2 = g[2], g3 = g[3];
      *(uint4*)&Kl[kr][kc]      = g0;
      *(uint4*)&Kl[kr][kc + 8]  = g1;
      *(uint4*)&Kl[kr][kc + 16] = g2;
      *(uint4*)&Kl[kr][kc + 24] = g3;
    }
    {  // V tile -> LDS transposed
      union { uint4 u4[4]; bf16 e[32]; } vb;
      const uint4* g = (const uint4*)(v + kvbase + (size_t)(kb + kr) * D_MODEL + kc);
      vb.u4[0] = g[0]; vb.u4[1] = g[1]; vb.u4[2] = g[2]; vb.u4[3] = g[3];
#pragma unroll
      for (int j = 0; j < 32; ++j) Vt[kc + j][kr] = vb.e[j];
    }
    __syncthreads();

    // S = Q @ K^T  (4 key sub-tiles of 16)
    v8f s[4];
#pragma unroll
    for (int nt = 0; nt < 4; ++nt) {
      s[nt] = zero8();
      const bf16* krw = &Kl[nt * 16 + lm][0];
#pragma unroll
      for (int c = 0; c < 4; ++c) {
        v16bf bfr = ldfrag(krw + c * 32 + hi * 8, krw + c * 32 + 16 + hi * 8);
        s[nt] = wmma_bf16(qf[c], bfr, s[nt]);
      }
    }

    // alibi + causal mask
#pragma unroll
    for (int nt = 0; nt < 4; ++nt) {
      const int key = kb + nt * 16 + lm;
#pragma unroll
      for (int i = 0; i < 8; ++i) {
        const int qr = q0 + wave * 16 + i + 8 * hi;
        float val;
        if (key > qr) val = NEG;
        else val = s[nt][i] + alibi[((size_t)h * SEQ + qr) * SEQ + key];
        s[nt][i] = val;
      }
    }

    // online softmax update
    float scal[8];
#pragma unroll
    for (int i = 0; i < 8; ++i) {
      float mx = fmaxf(fmaxf(s[0][i], s[1][i]), fmaxf(s[2][i], s[3][i]));
#pragma unroll
      for (int m = 1; m <= 8; m <<= 1) mx = fmaxf(mx, __shfl_xor(mx, m, 32));
      const float mn = fmaxf(mrow[i], mx);
      scal[i] = __expf(mrow[i] - mn);
      mrow[i] = mn;
      float rs = 0.f;
#pragma unroll
      for (int nt = 0; nt < 4; ++nt) {
        float p = __expf(s[nt][i] - mn);
        s[nt][i] = p;
        rs += p;
      }
#pragma unroll
      for (int m = 1; m <= 8; m <<= 1) rs += __shfl_xor(rs, m, 32);
      lsum[i] = lsum[i] * scal[i] + rs;
    }
#pragma unroll
    for (int ot = 0; ot < 8; ++ot)
#pragma unroll
      for (int i = 0; i < 8; ++i) o[ot][i] *= scal[i];

    // P (C layout) -> LDS -> A-fragment layout
#pragma unroll
    for (int nt = 0; nt < 4; ++nt)
#pragma unroll
      for (int i = 0; i < 8; ++i)
        Ps[wave][i + 8 * hi][nt * 16 + lm] = (bf16)s[nt][i];
    __syncthreads();

    // O += P @ V
#pragma unroll
    for (int c2 = 0; c2 < 2; ++c2) {
      const bf16* pr = &Ps[wave][lm][0];
      v16bf pf = ldfrag(pr + c2 * 32 + hi * 8, pr + c2 * 32 + 16 + hi * 8);
#pragma unroll
      for (int ot = 0; ot < 8; ++ot) {
        const bf16* vr = &Vt[ot * 16 + lm][0];
        v16bf bfr = ldfrag(vr + c2 * 32 + hi * 8, vr + c2 * 32 + 16 + hi * 8);
        o[ot] = wmma_bf16(pf, bfr, o[ot]);
      }
    }
  }

  // normalize and store (bf16, feeds O-projection GEMM)
  float inv[8];
#pragma unroll
  for (int i = 0; i < 8; ++i) inv[i] = 1.f / lsum[i];
#pragma unroll
  for (int ot = 0; ot < 8; ++ot)
#pragma unroll
    for (int i = 0; i < 8; ++i) {
      const int row = q0 + wave * 16 + i + 8 * hi;
      const int col = h * HEAD_DIM + ot * 16 + lm;
      out[((size_t)b * SEQ + row) * D_MODEL + col] = (bf16)(o[ot][i] * inv[i]);
    }
}

// ----------------------------------------------------------------- launch
extern "C" void kernel_launch(void* const* d_in, const int* in_sizes, int n_in,
                              void* d_out, int out_size, void* d_ws, size_t ws_size,
                              hipStream_t stream) {
  (void)in_sizes; (void)n_in; (void)out_size; (void)ws_size;
  const float* x      = (const float*)d_in[0];
  const float* alibi  = (const float*)d_in[1];
  const float* Wq     = (const float*)d_in[2];
  const float* Wk     = (const float*)d_in[3];
  const float* Wv     = (const float*)d_in[4];
  const float* Wo     = (const float*)d_in[5];
  const float* ln1g   = (const float*)d_in[6];
  const float* ln1b   = (const float*)d_in[7];
  const float* ln2g   = (const float*)d_in[8];
  const float* ln2b   = (const float*)d_in[9];
  const float* W1     = (const float*)d_in[10];
  const float* b1     = (const float*)d_in[11];
  const float* W2     = (const float*)d_in[12];
  const float* b2     = (const float*)d_in[13];
  float* outp = (float*)d_out;

  char* ws = (char*)d_ws;
  auto alloc = [&](size_t bytes) -> char* {
    char* p = ws;
    ws += (bytes + 255) & ~(size_t)255;
    return p;
  };
  const size_t e_dd = (size_t)D_MODEL * D_MODEL;
  const size_t e_df = (size_t)D_MODEL * D_FF;
  const size_t e_rd = (size_t)ROWS * D_MODEL;
  const size_t e_rf = (size_t)ROWS * D_FF;

  bf16* wq_t = (bf16*)alloc(e_dd * 2);
  bf16* wk_t = (bf16*)alloc(e_dd * 2);
  bf16* wv_t = (bf16*)alloc(e_dd * 2);
  bf16* wo_t = (bf16*)alloc(e_dd * 2);
  bf16* w1_t = (bf16*)alloc(e_df * 2);
  bf16* w2_t = (bf16*)alloc(e_df * 2);
  bf16* xn1  = (bf16*)alloc(e_rd * 2);
  bf16* qb   = (bf16*)alloc(e_rd * 2);
  bf16* kb   = (bf16*)alloc(e_rd * 2);
  bf16* vb   = (bf16*)alloc(e_rd * 2);
  bf16* attn = (bf16*)alloc(e_rd * 2);
  float* hb  = (float*)alloc(e_rd * 4);
  bf16* n2   = (bf16*)alloc(e_rd * 2);
  bf16* a1   = (bf16*)alloc(e_rf * 2);

  // weights -> bf16, transposed to [N][K] (coalesced LDS transpose)
  transpose_cvt_bf16<<<dim3(D_MODEL / 32, D_MODEL / 32), 256, 0, stream>>>(
      Wq, wq_t, D_MODEL, D_MODEL);
  transpose_cvt_bf16<<<dim3(D_MODEL / 32, D_MODEL / 32), 256, 0, stream>>>(
      Wk, wk_t, D_MODEL, D_MODEL);
  transpose_cvt_bf16<<<dim3(D_MODEL / 32, D_MODEL / 32), 256, 0, stream>>>(
      Wv, wv_t, D_MODEL, D_MODEL);
  transpose_cvt_bf16<<<dim3(D_MODEL / 32, D_MODEL / 32), 256, 0, stream>>>(
      Wo, wo_t, D_MODEL, D_MODEL);
  transpose_cvt_bf16<<<dim3(D_FF / 32, D_MODEL / 32), 256, 0, stream>>>(
      W1, w1_t, D_MODEL, D_FF);
  transpose_cvt_bf16<<<dim3(D_MODEL / 32, D_FF / 32), 256, 0, stream>>>(
      W2, w2_t, D_FF, D_MODEL);

  // LN1
  layernorm_bf16<<<ROWS, 256, 0, stream>>>(x, ln1g, ln1b, xn1);

  const dim3 gdd(D_MODEL / 128, ROWS / 128);
  const float qscale = 0.08838834764831845f;  // 1/sqrt(HEAD_DIM)
  gemm_bf16_wmma<<<gdd, 256, 0, stream>>>(xn1, wq_t, ROWS, D_MODEL, D_MODEL,
                                          qscale, nullptr, nullptr, 0, qb, nullptr);
  gemm_bf16_wmma<<<gdd, 256, 0, stream>>>(xn1, wk_t, ROWS, D_MODEL, D_MODEL,
                                          1.f, nullptr, nullptr, 0, kb, nullptr);
  gemm_bf16_wmma<<<gdd, 256, 0, stream>>>(xn1, wv_t, ROWS, D_MODEL, D_MODEL,
                                          1.f, nullptr, nullptr, 0, vb, nullptr);

  // attention
  flash_attn_wmma<<<dim3(SEQ / 128, N_HEADS, BATCH), 256, 0, stream>>>(
      qb, kb, vb, alibi, attn);

  // O projection + residual -> h (fp32)
  gemm_bf16_wmma<<<gdd, 256, 0, stream>>>(attn, wo_t, ROWS, D_MODEL, D_MODEL,
                                          1.f, nullptr, x, 0, nullptr, hb);
  // LN2
  layernorm_bf16<<<ROWS, 256, 0, stream>>>(hb, ln2g, ln2b, n2);
  // FFN1 + GELU
  gemm_bf16_wmma<<<dim3(D_FF / 128, ROWS / 128), 256, 0, stream>>>(
      n2, w1_t, ROWS, D_FF, D_MODEL, 1.f, b1, nullptr, 1, a1, nullptr);
  // FFN2 + bias + residual -> out (fp32)
  gemm_bf16_wmma<<<gdd, 256, 0, stream>>>(a1, w2_t, ROWS, D_MODEL, D_FF,
                                          1.f, b2, hb, 0, nullptr, outp);
}
